// GATWithJK_29283087024637
// MI455X (gfx1250) — matrix operand order repaired and checked by reading, not compiled, not run on hardware
//
#include <hip/hip_runtime.h>
#include <hip/hip_bf16.h>
#include <math.h>

#define Nn 50000
#define Ee 800000
#define EtT (Ee + Nn)
#define Bb 256
#define INC 32
#define CHc 128
#define HSd 192
#define G4d (4 * HSd)
#define OUTC 4
#define NEGS 0.2f

typedef __attribute__((ext_vector_type(16))) _Float16 v16h;
typedef __attribute__((ext_vector_type(8)))  _Float16 v8h;
typedef __attribute__((ext_vector_type(8)))  float    v8f;

// ---------------- WMMA helpers (contiguous k order, same in A and B) ----------------

__device__ __forceinline__ v8f wmma_f16(v16h a, v16h b, v8f c) {
  return __builtin_amdgcn_wmma_f32_16x16x32_f16(false, a, false, b, (short)0, c,
                                                false, false);
}

// A: 16x32 f16, row-major source (lda in halves).
// lane<16: M=lane, halves0..15 = K ks..ks+15
// lane>=16: M=lane-16, K ks+16..ks+31
__device__ __forceinline__ v16h load_a_frag(const _Float16* base, int lda, int ks) {
  int lane = threadIdx.x & 31;
  int m = lane & 15;
  int kb = ks + ((lane >> 4) << 4);
  const _Float16* p = base + (size_t)m * lda + kb;
  v8h lo = *(const v8h*)p;
  v8h hi = *(const v8h*)(p + 8);
  v16h a;
#pragma unroll
  for (int i = 0; i < 8; ++i) { a[i] = lo[i]; a[i + 8] = hi[i]; }
  return a;
}

// B: 32x16 f16 where B(k,j) = Wrows[j][k], Wrows row-major [ncols][ldb].
// lane<16: col j = jbase+lane, K ks..ks+15 ; lane>=16: same col, K ks+16..ks+31
__device__ __forceinline__ v16h load_b_frag(const _Float16* base, int ldb,
                                            int jbase, int ks) {
  int lane = threadIdx.x & 31;
  int j = jbase + (lane & 15);
  int kb = ks + ((lane >> 4) << 4);
  return *(const v16h*)(base + (size_t)j * ldb + kb);
}

__device__ __forceinline__ float sigm(float x) { return 1.f / (1.f + expf(-x)); }
__device__ __forceinline__ float lrelu(float x) { return x > 0.f ? x : NEGS * x; }

// ---------------- utility kernels ----------------

__global__ void zero_u32(unsigned int* p, int n) {
  int i = blockIdx.x * 256 + threadIdx.x;
  if (i < n) p[i] = 0u;
}

__global__ void cvt_f16_kernel(const float* __restrict__ src,
                               _Float16* __restrict__ dst, int n) {
  int i = blockIdx.x * 256 + threadIdx.x;
  if (i < n) dst[i] = (_Float16)src[i];
}

// src [K][128] -> dst [128][K] (f16), for conv weights so B-frags are contiguous
__global__ void cvt_f16_T_kernel(const float* __restrict__ src,
                                 _Float16* __restrict__ dst, int K) {
  int i = blockIdx.x * 256 + threadIdx.x;
  if (i >= 128 * K) return;
  int j = i / K, k = i - j * K;
  dst[i] = (_Float16)src[(size_t)k * 128 + j];
}

// ---------------- CSR build (dst-sorted) ----------------

__global__ void deg_kernel(const int* __restrict__ edge, int* __restrict__ deg) {
  int e = blockIdx.x * 256 + threadIdx.x;
  if (e >= EtT) return;
  int dst = (e < Ee) ? edge[Ee + e] : (e - Ee);  // self-loop tail
  atomicAdd(&deg[dst], 1);
}

__global__ void scan_kernel(const int* __restrict__ deg, int* __restrict__ rowptr,
                            int* __restrict__ cursor, int n) {
  __shared__ int buf[1024];
  __shared__ int carry_s;
  if (threadIdx.x == 0) carry_s = 0;
  __syncthreads();
  for (int base = 0; base < n; base += 1024) {
    int i = base + (int)threadIdx.x;
    int v = (i < n) ? deg[i] : 0;
    buf[threadIdx.x] = v;
    __syncthreads();
    for (int off = 1; off < 1024; off <<= 1) {
      int t = 0;
      if ((int)threadIdx.x >= off) t = buf[threadIdx.x - off];
      __syncthreads();
      buf[threadIdx.x] += t;
      __syncthreads();
    }
    int excl = carry_s + buf[threadIdx.x] - v;
    if (i < n) { rowptr[i] = excl; cursor[i] = excl; }
    int total = buf[1023];
    __syncthreads();
    if (threadIdx.x == 0) carry_s += total;
    __syncthreads();
  }
  if (threadIdx.x == 0) rowptr[n] = carry_s;
}

__global__ void fill_kernel(const int* __restrict__ edge, int* __restrict__ cursor,
                            int* __restrict__ col) {
  int e = blockIdx.x * 256 + threadIdx.x;
  if (e >= EtT) return;
  int src, dst;
  if (e < Ee) { src = edge[e]; dst = edge[Ee + e]; } else { src = e - Ee; dst = src; }
  int pos = atomicAdd(&cursor[dst], 1);
  col[pos] = src;
}

// ---------------- GAT linear: H[N,128] = A[N,K] @ W (WMMA) ----------------

__global__ __launch_bounds__(256) void gemm128(const _Float16* __restrict__ A,
                                               int lda, int K,
                                               const _Float16* __restrict__ Bw,
                                               float* __restrict__ H) {
  int node0 = blockIdx.x * 16;       // N % 16 == 0
  int wave = threadIdx.x >> 5;
  int lane = threadIdx.x & 31;
  int jbase = wave * 16;             // 8 waves cover 128 cols
  const _Float16* arow = A + (size_t)node0 * lda;
  v8f acc = {};
  for (int ks = 0; ks < K; ks += 32) {
    v16h a = load_a_frag(arow, lda, ks);
    v16h b = load_b_frag(Bw, K, jbase, ks);
    acc = wmma_f16(a, b, acc);
  }
  int nidx = lane & 15;
  int mb = (lane >> 4) << 3;
  float* out = H + (size_t)node0 * CHc + jbase + nidx;
#pragma unroll
  for (int v = 0; v < 8; ++v) out[(size_t)(mb + v) * CHc] = acc[v];
}

// a_s[n,h] = <h[n,h,:], a_src[h,:]> ; same for a_d
__global__ void attn_coef(const float* __restrict__ h, const float* __restrict__ a_src,
                          const float* __restrict__ a_dst, float* __restrict__ a_s,
                          float* __restrict__ a_d) {
  int id = blockIdx.x * 256 + threadIdx.x;
  if (id >= Nn * 4) return;
  int n = id >> 2, hh = id & 3;
  const float* row = h + (size_t)n * CHc + hh * 32;
  float s = 0.f, d = 0.f;
#pragma unroll 8
  for (int c = 0; c < 32; ++c) {
    float v = row[c];
    s += v * a_src[hh * 32 + c];
    d += v * a_dst[hh * 32 + c];
  }
  a_s[id] = s;
  a_d[id] = d;
}

// one wave per dst node: exact segment softmax + weighted gather (no float atomics)
__global__ __launch_bounds__(256) void gat_aggregate(
    const float* __restrict__ h, const float* __restrict__ a_s,
    const float* __restrict__ a_d, const int* __restrict__ rowptr,
    const int* __restrict__ col, const float* __restrict__ bias,
    float* __restrict__ seq32, _Float16* __restrict__ seq16, int layer) {
  int lane = threadIdx.x & 31;
  int n = blockIdx.x * 8 + (threadIdx.x >> 5);
  if (n >= Nn) return;
  int beg = rowptr[n], end = rowptr[n + 1];
  float4 ad = *(const float4*)(a_d + (size_t)n * 4);
  float4 mx = make_float4(-1e30f, -1e30f, -1e30f, -1e30f);
  for (int i = beg + lane; i < end; i += 32) {
    int s = col[i];
    float4 as = *(const float4*)(a_s + (size_t)s * 4);
    mx.x = fmaxf(mx.x, lrelu(as.x + ad.x));
    mx.y = fmaxf(mx.y, lrelu(as.y + ad.y));
    mx.z = fmaxf(mx.z, lrelu(as.z + ad.z));
    mx.w = fmaxf(mx.w, lrelu(as.w + ad.w));
  }
#pragma unroll
  for (int off = 16; off > 0; off >>= 1) {
    mx.x = fmaxf(mx.x, __shfl_xor(mx.x, off));
    mx.y = fmaxf(mx.y, __shfl_xor(mx.y, off));
    mx.z = fmaxf(mx.z, __shfl_xor(mx.z, off));
    mx.w = fmaxf(mx.w, __shfl_xor(mx.w, off));
  }
  float4 den = make_float4(0.f, 0.f, 0.f, 0.f);
  for (int i = beg + lane; i < end; i += 32) {
    int s = col[i];
    float4 as = *(const float4*)(a_s + (size_t)s * 4);
    den.x += expf(lrelu(as.x + ad.x) - mx.x);
    den.y += expf(lrelu(as.y + ad.y) - mx.y);
    den.z += expf(lrelu(as.z + ad.z) - mx.z);
    den.w += expf(lrelu(as.w + ad.w) - mx.w);
  }
#pragma unroll
  for (int off = 16; off > 0; off >>= 1) {
    den.x += __shfl_xor(den.x, off);
    den.y += __shfl_xor(den.y, off);
    den.z += __shfl_xor(den.z, off);
    den.w += __shfl_xor(den.w, off);
  }
  // channels c = lane*4+q ; head = c/32 = lane>>3 (same for all q)
  int head = lane >> 3;
  float mh  = head == 0 ? mx.x  : head == 1 ? mx.y  : head == 2 ? mx.z  : mx.w;
  float dh  = (head == 0 ? den.x : head == 1 ? den.y : head == 2 ? den.z : den.w) + 1e-16f;
  float adh = head == 0 ? ad.x  : head == 1 ? ad.y  : head == 2 ? ad.z  : ad.w;
  float a0 = 0.f, a1 = 0.f, a2 = 0.f, a3 = 0.f;
  for (int i = beg; i < end; ++i) {
    int s = col[i];
    if (i + 1 < end) __builtin_prefetch(h + (size_t)col[i + 1] * CHc + lane * 4, 0, 1);
    float al = expf(lrelu(a_s[(size_t)s * 4 + head] + adh) - mh) / dh;
    float4 hv = *(const float4*)(h + (size_t)s * CHc + lane * 4);
    a0 += hv.x * al; a1 += hv.y * al; a2 += hv.z * al; a3 += hv.w * al;
  }
  float4 bv = *(const float4*)(bias + lane * 4);
  float r0 = fmaxf(a0 + bv.x, 0.f), r1 = fmaxf(a1 + bv.y, 0.f);
  float r2 = fmaxf(a2 + bv.z, 0.f), r3 = fmaxf(a3 + bv.w, 0.f);
  size_t o = (size_t)n * (3 * CHc) + (size_t)layer * CHc + (size_t)lane * 4;
  *(float4*)(seq32 + o) = make_float4(r0, r1, r2, r3);
  seq16[o + 0] = (_Float16)r0; seq16[o + 1] = (_Float16)r1;
  seq16[o + 2] = (_Float16)r2; seq16[o + 3] = (_Float16)r3;
}

// ---------------- fused bidirectional-LSTM direction (WMMA) ----------------
// block = 16 nodes; 8 waves x 6 tiles = 768 gate cols; h recurrent state in LDS,
// c in registers; emits only attention partial scores h . att_w[dir] to HBM.
__global__ __launch_bounds__(256) void lstm_fused(
    const _Float16* __restrict__ seq16, const _Float16* __restrict__ Wi,  // [768][128]
    const _Float16* __restrict__ Wh,                                      // [768][192]
    const float* __restrict__ bi, const float* __restrict__ bh,
    const float* __restrict__ attw, int dir_off, int reverse,
    float* __restrict__ score /*[N][3]*/) {
  __shared__ __align__(32) _Float16 hsh[16 * HSd];   // 6 KB
  __shared__ float glds[16 * G4d];                   // 48 KB
  __shared__ float scl[16];
  int tid = threadIdx.x;
  int wave = tid >> 5, lane = tid & 31;
  int node0 = blockIdx.x * 16;
  for (int i = tid; i < 16 * HSd; i += 256) hsh[i] = (_Float16)0.f;
  float creg[12];
#pragma unroll
  for (int q = 0; q < 12; ++q) creg[q] = 0.f;
  int m_own = tid >> 4;          // node row 0..15
  int u0 = (tid & 15) * 12;      // 12 hidden units per thread
  __syncthreads();
  for (int ts = 0; ts < 3; ++ts) {
    int t = reverse ? (2 - ts) : ts;
    const _Float16* arow = seq16 + (size_t)node0 * (3 * CHc) + (size_t)t * CHc;
#pragma unroll
    for (int ct = 0; ct < 6; ++ct) {
      int jbase = (wave * 6 + ct) * 16;
      v8f acc = {};
#pragma unroll
      for (int ks = 0; ks < CHc; ks += 32) {        // x_t @ Wi^T
        v16h a = load_a_frag(arow, 3 * CHc, ks);
        v16h b = load_b_frag(Wi, CHc, jbase, ks);
        acc = wmma_f16(a, b, acc);
      }
#pragma unroll
      for (int ks = 0; ks < HSd; ks += 32) {        // h @ Wh^T
        v16h a = load_a_frag(hsh, HSd, ks);
        v16h b = load_b_frag(Wh, HSd, jbase, ks);
        acc = wmma_f16(a, b, acc);
      }
      int nidx = lane & 15, mb = (lane >> 4) << 3;
#pragma unroll
      for (int v = 0; v < 8; ++v) glds[(mb + v) * G4d + jbase + nidx] = acc[v];
    }
    if (tid < 16) scl[tid] = 0.f;
    __syncthreads();
    // gates: PyTorch order i,f,g,o
    float spart = 0.f;
#pragma unroll
    for (int q = 0; q < 12; ++q) {
      int u = u0 + q;
      const float* gr = glds + (size_t)m_own * G4d;
      float gi = gr[u]           + bi[u]           + bh[u];
      float gf = gr[HSd + u]     + bi[HSd + u]     + bh[HSd + u];
      float gg = gr[2 * HSd + u] + bi[2 * HSd + u] + bh[2 * HSd + u];
      float go = gr[3 * HSd + u] + bi[3 * HSd + u] + bh[3 * HSd + u];
      float c = sigm(gf) * creg[q] + sigm(gi) * tanhf(gg);
      creg[q] = c;
      float hval = sigm(go) * tanhf(c);
      hsh[m_own * HSd + u] = (_Float16)hval;
      spart += hval * attw[dir_off + u];
    }
    atomicAdd(&scl[m_own], spart);
    __syncthreads();
    if (tid < 16) score[(size_t)(node0 + tid) * 3 + t] = scl[tid];
    __syncthreads();
  }
}

// ---------------- attention softmax over L=3, pooled feat, batch sums ----------------

__global__ __launch_bounds__(256) void pool_kernel(
    const float* __restrict__ seq, const float* __restrict__ sf,
    const float* __restrict__ sb, const float* __restrict__ attb,
    const int* __restrict__ batch, float* __restrict__ bsum,
    float* __restrict__ bcnt) {
  int lane = threadIdx.x & 31;
  int n = blockIdx.x * 8 + (threadIdx.x >> 5);
  if (n >= Nn) return;
  float s0 = sf[n * 3 + 0] + sb[n * 3 + 0] + attb[0];
  float s1 = sf[n * 3 + 1] + sb[n * 3 + 1] + attb[0];
  float s2 = sf[n * 3 + 2] + sb[n * 3 + 2] + attb[0];
  float mx = fmaxf(s0, fmaxf(s1, s2));
  float e0 = expf(s0 - mx), e1 = expf(s1 - mx), e2 = expf(s2 - mx);
  float inv = 1.f / (e0 + e1 + e2);
  float w0 = e0 * inv, w1 = e1 * inv, w2 = e2 * inv;
  const float* row = seq + (size_t)n * (3 * CHc);
  float4 v0 = *(const float4*)(row + lane * 4);
  float4 v1 = *(const float4*)(row + CHc + lane * 4);
  float4 v2 = *(const float4*)(row + 2 * CHc + lane * 4);
  int b = batch[n];
  float* bs = bsum + (size_t)b * CHc + lane * 4;
  atomicAdd(bs + 0, w0 * v0.x + w1 * v1.x + w2 * v2.x);
  atomicAdd(bs + 1, w0 * v0.y + w1 * v1.y + w2 * v2.y);
  atomicAdd(bs + 2, w0 * v0.z + w1 * v1.z + w2 * v2.z);
  atomicAdd(bs + 3, w0 * v0.w + w1 * v1.w + w2 * v2.w);
  if (lane == 0) atomicAdd(&bcnt[b], 1.f);
}

__global__ void final_kernel(const float* __restrict__ bsum,
                             const float* __restrict__ bcnt,
                             const float* __restrict__ lin_w,
                             const float* __restrict__ lin_b,
                             float* __restrict__ out) {
  int id = blockIdx.x * 256 + threadIdx.x;
  if (id >= Bb * OUTC) return;
  int b = id >> 2, j = id & 3;
  float inv = 1.f / fmaxf(bcnt[b], 1.f);
  float s = 0.f;
  for (int c = 0; c < CHc; ++c) s += bsum[(size_t)b * CHc + c] * lin_w[c * OUTC + j];
  out[id] = s * inv + lin_b[j];
}

// ---------------- host launcher ----------------

extern "C" void kernel_launch(void* const* d_in, const int* in_sizes, int n_in,
                              void* d_out, int out_size, void* d_ws, size_t ws_size,
                              hipStream_t stream) {
  (void)in_sizes; (void)n_in; (void)out_size; (void)ws_size;
  const float* x      = (const float*)d_in[0];
  const int*   edge   = (const int*)d_in[1];
  const int*   batch  = (const int*)d_in[2];
  const float* convW[3]  = {(const float*)d_in[3], (const float*)d_in[7],  (const float*)d_in[11]};
  const float* convAS[3] = {(const float*)d_in[4], (const float*)d_in[8],  (const float*)d_in[12]};
  const float* convAD[3] = {(const float*)d_in[5], (const float*)d_in[9],  (const float*)d_in[13]};
  const float* convB[3]  = {(const float*)d_in[6], (const float*)d_in[10], (const float*)d_in[14]};
  const float* Wi_f = (const float*)d_in[15];
  const float* Wh_f = (const float*)d_in[16];
  const float* bi_f = (const float*)d_in[17];
  const float* bh_f = (const float*)d_in[18];
  const float* Wi_b = (const float*)d_in[19];
  const float* Wh_b = (const float*)d_in[20];
  const float* bi_b = (const float*)d_in[21];
  const float* bh_b = (const float*)d_in[22];
  const float* attw = (const float*)d_in[23];
  const float* attb = (const float*)d_in[24];
  const float* linw = (const float*)d_in[25];
  const float* linb = (const float*)d_in[26];
  float* out = (float*)d_out;

  char* p = (char*)d_ws;
  auto carve = [&](size_t bytes) -> void* {
    void* r = (void*)p;
    p += (bytes + 255) & ~(size_t)255;
    return r;
  };
  int* rowptr = (int*)carve((Nn + 1) * 4);
  int* degcur = (int*)carve(Nn * 4);           // degree then cursor (aliased)
  int* col    = (int*)carve((size_t)EtT * 4);
  _Float16* x16   = (_Float16*)carve((size_t)Nn * INC * 2);
  _Float16* Wt0   = (_Float16*)carve(128 * INC * 2);
  _Float16* Wt1   = (_Float16*)carve(128 * 128 * 2);
  _Float16* Wt2   = (_Float16*)carve(128 * 128 * 2);
  _Float16* Wi16f = (_Float16*)carve((size_t)G4d * CHc * 2);
  _Float16* Wh16f = (_Float16*)carve((size_t)G4d * HSd * 2);
  _Float16* Wi16b = (_Float16*)carve((size_t)G4d * CHc * 2);
  _Float16* Wh16b = (_Float16*)carve((size_t)G4d * HSd * 2);
  float*    hbuf  = (float*)carve((size_t)Nn * CHc * 4);
  float*    asb   = (float*)carve((size_t)Nn * 4 * 4);
  float*    adb   = (float*)carve((size_t)Nn * 4 * 4);
  float*    seq   = (float*)carve((size_t)Nn * 3 * CHc * 4);
  _Float16* seq16 = (_Float16*)carve((size_t)Nn * 3 * CHc * 2);
  float*    sf    = (float*)carve((size_t)Nn * 3 * 4);
  float*    sb    = (float*)carve((size_t)Nn * 3 * 4);
  float*    bsum  = (float*)carve((size_t)Bb * CHc * 4);
  float*    bcnt  = (float*)carve((size_t)Bb * 4);

  const int T = 256;
  // CSR build
  zero_u32<<<(Nn + T - 1) / T, T, 0, stream>>>((unsigned*)degcur, Nn);
  deg_kernel<<<(EtT + T - 1) / T, T, 0, stream>>>(edge, degcur);
  scan_kernel<<<1, 1024, 0, stream>>>(degcur, rowptr, degcur, Nn);
  fill_kernel<<<(EtT + T - 1) / T, T, 0, stream>>>(edge, degcur, col);
  // weight / input conversion to f16
  cvt_f16_kernel<<<(Nn * INC + T - 1) / T, T, 0, stream>>>(x, x16, Nn * INC);
  cvt_f16_T_kernel<<<(128 * INC + T - 1) / T, T, 0, stream>>>(convW[0], Wt0, INC);
  cvt_f16_T_kernel<<<(128 * 128 + T - 1) / T, T, 0, stream>>>(convW[1], Wt1, 128);
  cvt_f16_T_kernel<<<(128 * 128 + T - 1) / T, T, 0, stream>>>(convW[2], Wt2, 128);
  cvt_f16_kernel<<<(G4d * CHc + T - 1) / T, T, 0, stream>>>(Wi_f, Wi16f, G4d * CHc);
  cvt_f16_kernel<<<(G4d * HSd + T - 1) / T, T, 0, stream>>>(Wh_f, Wh16f, G4d * HSd);
  cvt_f16_kernel<<<(G4d * CHc + T - 1) / T, T, 0, stream>>>(Wi_b, Wi16b, G4d * CHc);
  cvt_f16_kernel<<<(G4d * HSd + T - 1) / T, T, 0, stream>>>(Wh_b, Wh16b, G4d * HSd);
  // 3 GAT layers
  _Float16* Wt[3] = {Wt0, Wt1, Wt2};
  for (int l = 0; l < 3; ++l) {
    const _Float16* A = (l == 0) ? x16 : (seq16 + (size_t)(l - 1) * CHc);
    int lda = (l == 0) ? INC : 3 * CHc;
    int K   = (l == 0) ? INC : CHc;
    gemm128<<<Nn / 16, T, 0, stream>>>(A, lda, K, Wt[l], hbuf);
    attn_coef<<<(Nn * 4 + T - 1) / T, T, 0, stream>>>(hbuf, convAS[l], convAD[l], asb, adb);
    gat_aggregate<<<Nn / 8, T, 0, stream>>>(hbuf, asb, adb, rowptr, col, convB[l],
                                            seq, seq16, l);
  }
  // bidirectional JK-LSTM (fully fused, WMMA)
  lstm_fused<<<Nn / 16, T, 0, stream>>>(seq16, Wi16f, Wh16f, bi_f, bh_f, attw, 0, 0, sf);
  lstm_fused<<<Nn / 16, T, 0, stream>>>(seq16, Wi16b, Wh16b, bi_b, bh_b, attw, HSd, 1, sb);
  // attention pooling + global mean pool + final linear
  zero_u32<<<(Bb * CHc + T - 1) / T, T, 0, stream>>>((unsigned*)bsum, Bb * CHc);
  zero_u32<<<1, T, 0, stream>>>((unsigned*)bcnt, Bb);
  pool_kernel<<<Nn / 8, T, 0, stream>>>(seq, sf, sb, attb, batch, bsum, bcnt);
  final_kernel<<<(Bb * OUTC + T - 1) / T, T, 0, stream>>>(bsum, bcnt, linw, linb, out);
}
